// TriangularSelfAttention_52810917871758
// MI455X (gfx1250) — matrix operand order repaired and checked by reading, not compiled, not run on hardware
//
#include <hip/hip_runtime.h>

typedef __attribute__((ext_vector_type(16))) __bf16 v16bf;
typedef __attribute__((ext_vector_type(8)))  float  v8f;

#define WMMA(a, b, c) __builtin_amdgcn_wmma_f32_16x16x32_bf16(false, (a), false, (b), (short)0, (c), false, false)

#define XS_STR 136   // padded row stride (ushorts) for 256x128 x-slice
#define KQ_STR 72    // padded row stride for 256x64 K / Q
#define VT_STR 264   // padded row stride for 64x256 V^T
#define SMEM_BYTES 326656

__device__ __forceinline__ unsigned short f2bf(float f) {
  union { __bf16 b; unsigned short u; } x;
  x.b = (__bf16)f;                       // native v_cvt f32->bf16 (RNE)
  return x.u;
}
__device__ __forceinline__ __bf16 bfbits(unsigned short s) {
  union { unsigned short u; __bf16 b; } x; x.u = s; return x.b;
}

// A operand: lane holds row (lane&15); element j -> K = cbase+2*hiA+j
__device__ __forceinline__ v16bf fragA(const unsigned short* row, int cbase, int hiA) {
  v16bf a;
#pragma unroll
  for (int j = 0; j < 8; ++j) {
    a[j]     = bfbits(row[cbase + 2 * hiA + j]);
    a[j + 8] = bfbits(row[cbase + 2 * hiA + 8 + j]);
  }
  return a;
}
// B operand: lane holds column (lane&15) of B, i.e. a row of B^T; K = kbase + j (kbase includes hiB)
__device__ __forceinline__ v16bf fragB(const unsigned short* rowT, int kbase) {
  v16bf b;
#pragma unroll
  for (int j = 0; j < 16; ++j) b[j] = bfbits(rowT[kbase + j]);
  return b;
}

// ---------------- Pass A: LayerNorm (fp32 -> bf16) + pair_bias (+mask bias) ----------------
__global__ void ln_bias_kernel(const float* __restrict__ pair, const float* __restrict__ mask,
                               const float* __restrict__ gamma, const float* __restrict__ beta,
                               const float* __restrict__ Wb,
                               unsigned short* __restrict__ xbf, float* __restrict__ pbias) {
  const int lane = threadIdx.x & 31;
  const int wv = threadIdx.x >> 5;
  const int p = blockIdx.x * 8 + wv;            // pixel (q*256+v), 0..65535
  const float4 x = ((const float4*)(pair + (size_t)p * 128))[lane];
  float s  = x.x + x.y + x.z + x.w;
  float ss = x.x * x.x + x.y * x.y + x.z * x.z + x.w * x.w;
#pragma unroll
  for (int off = 1; off < 32; off <<= 1) { s += __shfl_xor(s, off, 32); ss += __shfl_xor(ss, off, 32); }
  const float mu = s * (1.0f / 128.0f);
  const float var = ss * (1.0f / 128.0f) - mu * mu;
  const float rs = rsqrtf(var + 1e-5f);
  const float4 g = ((const float4*)gamma)[lane];
  const float4 b = ((const float4*)beta)[lane];
  const float y0 = (x.x - mu) * rs * g.x + b.x;
  const float y1 = (x.y - mu) * rs * g.y + b.y;
  const float y2 = (x.z - mu) * rs * g.z + b.z;
  const float y3 = (x.w - mu) * rs * g.w + b.w;
  ushort4 o; o.x = f2bf(y0); o.y = f2bf(y1); o.z = f2bf(y2); o.w = f2bf(y3);
  ((ushort4*)(xbf + (size_t)p * 128))[lane] = o;

  const int d0 = lane * 4;
  float acc[4];
#pragma unroll
  for (int h = 0; h < 4; ++h)
    acc[h] = y0 * Wb[(d0 + 0) * 4 + h] + y1 * Wb[(d0 + 1) * 4 + h] +
             y2 * Wb[(d0 + 2) * 4 + h] + y3 * Wb[(d0 + 3) * 4 + h];
#pragma unroll
  for (int h = 0; h < 4; ++h)
#pragma unroll
    for (int off = 1; off < 32; off <<= 1) acc[h] += __shfl_xor(acc[h], off, 32);
  if (lane == 0) {
    const float mb = 1e9f * (mask[p] - 1.0f);
#pragma unroll
    for (int h = 0; h < 4; ++h) pbias[h * 65536 + p] = acc[h] + mb;
  }
}

// ---------------- Pass B: weights -> bf16, transposed to B^T-friendly layout ----------------
__global__ void wprep_kernel(const float* __restrict__ Wq, const float* __restrict__ Wk,
                             const float* __restrict__ Wv, const float* __restrict__ Wu,
                             const float* __restrict__ Wo,
                             unsigned short* __restrict__ wqt, unsigned short* __restrict__ wkt,
                             unsigned short* __restrict__ wvt, unsigned short* __restrict__ wut,
                             unsigned short* __restrict__ wot) {
  const int idx = blockIdx.x * 256 + threadIdx.x;      // 0 .. 163839
  if (idx >= 5 * 32768) return;
  const int m = idx >> 15;
  const int r = idx & 32767;
  if (m < 4) {                                         // [h][cout][din] = W[d][h*64+c]
    const int h = r >> 13;
    const int c = (r >> 7) & 63;
    const int d = r & 127;
    const float* W = (m == 0) ? Wq : (m == 1) ? Wk : (m == 2) ? Wv : Wu;
    unsigned short* T = (m == 0) ? wqt : (m == 1) ? wkt : (m == 2) ? wvt : wut;
    T[r] = f2bf(W[d * 256 + h * 64 + c]);
  } else {                                             // WoT[h][d][c] = Wo[h*64+c][d]
    const int h = r >> 13;
    const int d = (r >> 6) & 127;
    const int c = r & 63;
    wot[r] = f2bf(Wo[(h * 64 + c) * 128 + d]);
  }
}

// ---------------- Pass C: fused per-s attention ----------------
__launch_bounds__(256, 1)
__global__ void attn_kernel(const unsigned short* __restrict__ xbf,
                            const float* __restrict__ pbias,
                            const unsigned short* __restrict__ wqt,
                            const unsigned short* __restrict__ wkt,
                            const unsigned short* __restrict__ wvt,
                            const unsigned short* __restrict__ wut,
                            const unsigned short* __restrict__ wot,
                            const float* __restrict__ bu,
                            const float* __restrict__ bo,
                            const float* __restrict__ mask,
                            float* __restrict__ out) {
  extern __shared__ char smem[];
  unsigned short* XS = (unsigned short*)(smem);            // [256][136] bf16 x-slice (LDS offset 0)
  unsigned short* KM = (unsigned short*)(smem + 69632);    // [256][72]  K
  unsigned short* QM = (unsigned short*)(smem + 106496);   // [256][72]  Q (scaled)
  unsigned short* VT = (unsigned short*)(smem + 143360);   // [64][264]  V^T
  float*          AC = (float*)(smem + 177152);            // [256][128] f32 head accumulator
  unsigned short* SC = (unsigned short*)(smem + 308224);   // 8 waves x 1152 ushorts scratch

  const int tid = threadIdx.x;
  const int lane = tid & 31;
  const int wv = tid >> 5;
  const int l15 = lane & 15;
  const int hiA = (lane >> 4) << 3;   // 0 or 8
  const int hiB = (lane >> 4) << 4;   // 0 or 16
  const int s = blockIdx.x;

  // Stage x slice (bf16) into padded LDS via CDNA5 async LDS-DMA; zero accumulator meanwhile
#if defined(__gfx1250__)
  for (int i = tid; i < 4096; i += 256) {
    const int row = i >> 4, seg = i & 15;
    const unsigned ldsoff = (unsigned)(row * (XS_STR * 2) + seg * 16);  // bytes from LDS base
    const unsigned long long g =
        (unsigned long long)(xbf + (size_t)(s * 256 + row) * 128 + seg * 8);
    asm volatile("global_load_async_to_lds_b128 %0, %1, off"
                 :: "v"(ldsoff), "v"(g) : "memory");
  }
  for (int i = tid; i < 256 * 128; i += 256) AC[i] = 0.0f;
  asm volatile("s_wait_asynccnt 0x0" ::: "memory");
#else
  for (int i = tid; i < 4096; i += 256) {
    const int row = i >> 4, seg = i & 15;
    ((uint4*)(XS + row * XS_STR))[seg] =
        ((const uint4*)(xbf + (size_t)(s * 256 + row) * 128))[seg];
  }
  for (int i = tid; i < 256 * 128; i += 256) AC[i] = 0.0f;
#endif
  __syncthreads();

  unsigned short* scr = SC + wv * 1152;

  for (int h = 0; h < 4; ++h) {
    // ---- projections K, Q (x0.125), V (transposed) ----
    for (int tt = 0; tt < 2; ++tt) {
      const int rowbase = (wv + tt * 8) * 16;
      const unsigned short* xrow = XS + (rowbase + l15) * XS_STR;
      v16bf ax[4];
#pragma unroll
      for (int kc = 0; kc < 4; ++kc) ax[kc] = fragA(xrow, kc * 32, hiA);
#pragma unroll
      for (int ct = 0; ct < 4; ++ct) {
        const unsigned short* wq = wqt + (h * 64 + ct * 16 + l15) * 128;
        const unsigned short* wk = wkt + (h * 64 + ct * 16 + l15) * 128;
        const unsigned short* wvp = wvt + (h * 64 + ct * 16 + l15) * 128;
        v8f aq = {0, 0, 0, 0, 0, 0, 0, 0};
        v8f ak = aq, av = aq;
#pragma unroll
        for (int kc = 0; kc < 4; ++kc) {
          aq = WMMA(ax[kc], fragB(wq, kc * 32 + hiB), aq);
          ak = WMMA(ax[kc], fragB(wk, kc * 32 + hiB), ak);
          av = WMMA(ax[kc], fragB(wvp, kc * 32 + hiB), av);
        }
#pragma unroll
        for (int r = 0; r < 8; ++r) {
          const int orow = rowbase + r + hiA;
          const int oc = ct * 16 + l15;
          QM[orow * KQ_STR + oc] = f2bf(aq[r] * 0.125f);
          KM[orow * KQ_STR + oc] = f2bf(ak[r]);
          VT[oc * VT_STR + orow] = f2bf(av[r]);
        }
      }
    }
    __syncthreads();

    // ---- attention: per wave, two 16-row q-tiles ----
    for (int tt = 0; tt < 2; ++tt) {
      const int qbase = (wv + tt * 8) * 16;
      const unsigned short* krow = KM + (qbase + l15) * KQ_STR;
      const v16bf ka0 = fragA(krow, 0, hiA);
      const v16bf ka1 = fragA(krow, 32, hiA);
      v8f S[16];
#pragma unroll
      for (int vt = 0; vt < 16; ++vt) {
        const unsigned short* qrow = QM + (vt * 16 + l15) * KQ_STR;
        v8f acc = {0, 0, 0, 0, 0, 0, 0, 0};
        acc = WMMA(ka0, fragB(qrow, 0 + hiB), acc);
        acc = WMMA(ka1, fragB(qrow, 32 + hiB), acc);
        const float* pb = pbias + h * 65536 + (qbase + hiA) * 256 + vt * 16 + l15;
#pragma unroll
        for (int r = 0; r < 8; ++r) acc[r] += pb[r * 256];
        S[vt] = acc;
      }
      // softmax over columns (vi); row lives across the 16 lanes of each half-wave
      float mx[8], sm[8], rinv[8];
#pragma unroll
      for (int r = 0; r < 8; ++r) mx[r] = S[0][r];
#pragma unroll
      for (int vt = 1; vt < 16; ++vt)
#pragma unroll
        for (int r = 0; r < 8; ++r) mx[r] = fmaxf(mx[r], S[vt][r]);
#pragma unroll
      for (int r = 0; r < 8; ++r)
#pragma unroll
        for (int off = 1; off < 16; off <<= 1) mx[r] = fmaxf(mx[r], __shfl_xor(mx[r], off, 32));
#pragma unroll
      for (int r = 0; r < 8; ++r) sm[r] = 0.0f;
#pragma unroll
      for (int vt = 0; vt < 16; ++vt)
#pragma unroll
        for (int r = 0; r < 8; ++r) { const float p = __expf(S[vt][r] - mx[r]); S[vt][r] = p; sm[r] += p; }
#pragma unroll
      for (int r = 0; r < 8; ++r)
#pragma unroll
        for (int off = 1; off < 16; off <<= 1) sm[r] += __shfl_xor(sm[r], off, 32);
#pragma unroll
      for (int r = 0; r < 8; ++r) rinv[r] = 1.0f / sm[r];

      // O = P @ V (unnormalized), 32-column chunks via per-wave LDS transpose bounce
      v8f O[4];
#pragma unroll
      for (int ct = 0; ct < 4; ++ct) { v8f z = {0, 0, 0, 0, 0, 0, 0, 0}; O[ct] = z; }
      for (int ch = 0; ch < 8; ++ch) {
#pragma unroll
        for (int r = 0; r < 8; ++r) {
          scr[(r + hiA) * 34 + l15]      = f2bf(S[2 * ch][r]);
          scr[(r + hiA) * 34 + 16 + l15] = f2bf(S[2 * ch + 1][r]);
        }
        __asm__ volatile("" ::: "memory");
        const v16bf pa = fragA(scr + l15 * 34, 0, hiA);
        __asm__ volatile("" ::: "memory");
#pragma unroll
        for (int ct = 0; ct < 4; ++ct)
          O[ct] = WMMA(pa, fragB(VT + (ct * 16 + l15) * VT_STR, ch * 32 + hiB), O[ct]);
      }

      // gate: G = (O/sum) * sigmoid(x@Wu + bu)
      const unsigned short* xrow = XS + (qbase + l15) * XS_STR;
      v16bf ax[4];
#pragma unroll
      for (int kc = 0; kc < 4; ++kc) ax[kc] = fragA(xrow, kc * 32, hiA);
#pragma unroll
      for (int ct = 0; ct < 4; ++ct) {
        const unsigned short* wu = wut + (h * 64 + ct * 16 + l15) * 128;
        v8f u = {0, 0, 0, 0, 0, 0, 0, 0};
#pragma unroll
        for (int kc = 0; kc < 4; ++kc) u = WMMA(ax[kc], fragB(wu, kc * 32 + hiB), u);
        const float bc = bu[h * 64 + ct * 16 + l15];
#pragma unroll
        for (int r = 0; r < 8; ++r) {
          const float sg = 1.0f / (1.0f + __expf(-(u[r] + bc)));
          scr[(r + hiA) * 68 + ct * 16 + l15] = f2bf(O[ct][r] * rinv[r] * sg);
        }
      }
      __asm__ volatile("" ::: "memory");
      const v16bf ga0 = fragA(scr + l15 * 68, 0, hiA);
      const v16bf ga1 = fragA(scr + l15 * 68, 32, hiA);
      __asm__ volatile("" ::: "memory");
      // Y[qtile] += G @ Wo_h  (accumulate across heads in LDS f32)
#pragma unroll
      for (int dt = 0; dt < 8; ++dt) {
        float* ac = AC + (qbase + hiA) * 128 + dt * 16 + l15;
        v8f cfr;
#pragma unroll
        for (int r = 0; r < 8; ++r) cfr[r] = ac[r * 128];
        const unsigned short* wo = wot + (h * 128 + dt * 16 + l15) * 64;
        cfr = WMMA(ga0, fragB(wo, 0 + hiB), cfr);
        cfr = WMMA(ga1, fragB(wo, 32 + hiB), cfr);
#pragma unroll
        for (int r = 0; r < 8; ++r) ac[r * 128] = cfr[r];
      }
    }
    __syncthreads();
  }

  // epilogue: + bo, * mask[s, q], store fp32
  for (int i = tid; i < 256 * 128; i += 256) {
    const int row = i >> 7, d = i & 127;
    out[(size_t)(s * 256 + row) * 128 + d] = (AC[i] + bo[d]) * mask[s * 256 + row];
  }
}

extern "C" void kernel_launch(void* const* d_in, const int* in_sizes, int n_in,
                              void* d_out, int out_size, void* d_ws, size_t ws_size,
                              hipStream_t stream) {
  const float* pair  = (const float*)d_in[0];
  const float* mask  = (const float*)d_in[1];
  const float* gamma = (const float*)d_in[2];
  const float* beta  = (const float*)d_in[3];
  const float* Wq    = (const float*)d_in[4];
  const float* Wk    = (const float*)d_in[5];
  const float* Wv    = (const float*)d_in[6];
  const float* Wb    = (const float*)d_in[7];
  const float* Wu    = (const float*)d_in[8];
  const float* bu    = (const float*)d_in[9];
  const float* Wo    = (const float*)d_in[10];
  const float* bo    = (const float*)d_in[11];
  float* out = (float*)d_out;

  char* ws = (char*)d_ws;
  unsigned short* xbf   = (unsigned short*)(ws);             // 16 MB bf16 LN(x)
  float*          pbias = (float*)(ws + 16777216);           // 1 MB [h][q][v]
  unsigned short* wqt   = (unsigned short*)(ws + 17825792);  // 64 KB each
  unsigned short* wkt   = (unsigned short*)(ws + 17891328);
  unsigned short* wvt   = (unsigned short*)(ws + 17956864);
  unsigned short* wut   = (unsigned short*)(ws + 18022400);
  unsigned short* wot   = (unsigned short*)(ws + 18087936);

  hipLaunchKernelGGL(ln_bias_kernel, dim3(8192), dim3(256), 0, stream,
                     pair, mask, gamma, beta, Wb, xbf, pbias);
  hipLaunchKernelGGL(wprep_kernel, dim3(640), dim3(256), 0, stream,
                     Wq, Wk, Wv, Wu, Wo, wqt, wkt, wvt, wut, wot);
  hipFuncSetAttribute(reinterpret_cast<const void*>(attn_kernel),
                      hipFuncAttributeMaxDynamicSharedMemorySize, SMEM_BYTES);
  hipLaunchKernelGGL(attn_kernel, dim3(256), dim3(256), SMEM_BYTES, stream,
                     xbf, pbias, wqt, wkt, wvt, wut, wot, bu, bo, mask, out);
}
